// SparseMoeWrapper_6897717477960
// MI455X (gfx1250) — matrix operand order repaired and checked by reading, not compiled, not run on hardware
//
#include <hip/hip_runtime.h>

// ---------------- problem constants (from reference) ----------------
#define TOK  4096   // B*S = 2*2048
#define HID  1024   // H
#define FF   4096   // F
#define NE   8      // experts (top_k fixed at 2)

typedef __attribute__((ext_vector_type(16))) __bf16 v16bf;
typedef __attribute__((ext_vector_type(8)))  __bf16 v8bf;
typedef __attribute__((ext_vector_type(8)))  float  v8f;
typedef __attribute__((ext_vector_type(4)))  float  v4f;

// ---- CDNA5 async copy to LDS (guarded; fallback = load + ds_store) ----
#if defined(__has_builtin)
#  if __has_builtin(__builtin_amdgcn_global_load_async_to_lds_b128)
#    define HAVE_ASYNC_B128 1
#  endif
#  if __has_builtin(__builtin_amdgcn_s_wait_asynccnt)
#    define HAVE_WAIT_ASYNC 1
#  endif
#endif
#ifndef HAVE_ASYNC_B128
#  define HAVE_ASYNC_B128 0
#endif
#ifndef HAVE_WAIT_ASYNC
#  define HAVE_WAIT_ASYNC 0
#endif

#if HAVE_ASYNC_B128
// exact param shape from clang diagnostic: int __vector(4) in AS1 (global) / AS3 (LDS)
typedef int v4i __attribute__((vector_size(16)));
typedef __attribute__((address_space(1))) v4i* as1_v4i;
typedef __attribute__((address_space(3))) v4i* as3_v4i;
#define ASYNC_CP_B128(gsrc, lds_dst)                                            \
    __builtin_amdgcn_global_load_async_to_lds_b128(                             \
        (as1_v4i)(uintptr_t)(gsrc),                                             \
        (as3_v4i)(uint32_t)(uintptr_t)(lds_dst), 0, 0)
#endif

static __device__ __forceinline__ v8bf cvt8(const v4f f0, const v4f f1) {
    v8bf b;
    b[0] = (__bf16)f0[0]; b[1] = (__bf16)f0[1];
    b[2] = (__bf16)f0[2]; b[3] = (__bf16)f0[3];
    b[4] = (__bf16)f1[0]; b[5] = (__bf16)f1[1];
    b[6] = (__bf16)f1[2]; b[7] = (__bf16)f1[3];
    return b;
}

// ---------------- zero-init helpers ----------------
__global__ void moe_zero(float* __restrict__ p, int n) {
    int i = blockIdx.x * 256 + threadIdx.x;
    if (i < n) p[i] = 0.0f;
}
__global__ void moe_zero_counts(int* __restrict__ p) {
    if (threadIdx.x < NE) p[threadIdx.x] = 0;
}

// ---------------- router: logits, softmax, top-2, gather lists ----------------
// block = 256 threads = 8 waves; one wave per token
__global__ __launch_bounds__(256) void moe_router(
    const float* __restrict__ x,      // [TOK][HID]
    const float* __restrict__ gwm,    // [NE][HID]
    int*   __restrict__ counts,       // [NE]
    int*   __restrict__ idx,          // [NE][TOK]
    float* __restrict__ gathw,        // [NE][TOK]
    float* __restrict__ logits_out)   // [TOK][NE] (2nd tuple output)
{
    const int lane = threadIdx.x & 31;
    const int wid  = threadIdx.x >> 5;
    const int t    = blockIdx.x * 8 + wid;

    float acc[NE];
#pragma unroll
    for (int e = 0; e < NE; ++e) acc[e] = 0.0f;

    for (int h = lane; h < HID; h += 32) {
        float xv = x[(size_t)t * HID + h];
#pragma unroll
        for (int e = 0; e < NE; ++e)
            acc[e] = fmaf(xv, gwm[(size_t)e * HID + h], acc[e]);
    }
#pragma unroll
    for (int e = 0; e < NE; ++e) {
        float v = acc[e];
#pragma unroll
        for (int off = 16; off > 0; off >>= 1) v += __shfl_down(v, off, 32);
        acc[e] = v;
    }
    if (lane == 0) {
        float mx = acc[0];
#pragma unroll
        for (int e = 1; e < NE; ++e) mx = fmaxf(mx, acc[e]);
        float p[NE], s = 0.0f;
#pragma unroll
        for (int e = 0; e < NE; ++e) { p[e] = __expf(acc[e] - mx); s += p[e]; }
        float inv = 1.0f / s;
#pragma unroll
        for (int e = 0; e < NE; ++e) p[e] *= inv;
        // top-2 (first index wins ties, matching top_k order)
        int i0 = 0;
#pragma unroll
        for (int e = 1; e < NE; ++e) if (p[e] > p[i0]) i0 = e;
        int i1 = (i0 == 0) ? 1 : 0;
#pragma unroll
        for (int e = 0; e < NE; ++e) if (e != i0 && p[e] > p[i1]) i1 = e;
        float wsum = p[i0] + p[i1];
        float w0 = p[i0] / wsum, w1 = p[i1] / wsum;
        // gather-list append (slot order nondeterministic; results are not:
        // each row's GEMM result is independent of its slot)
        int s0 = atomicAdd(&counts[i0], 1);
        idx[(size_t)i0 * TOK + s0]   = t;
        gathw[(size_t)i0 * TOK + s0] = w0;
        int s1 = atomicAdd(&counts[i1], 1);
        idx[(size_t)i1 * TOK + s1]   = t;
        gathw[(size_t)i1 * TOK + s1] = w1;
#pragma unroll
        for (int e = 0; e < NE; ++e)
            logits_out[(size_t)t * NE + e] = acc[e];
    }
}

// ---------------- expert GEMM 1 (gathered): inter = silu(Xg@w1^T) * (Xg@w3^T) ----------------
// block tile 128(M over slots) x 64(N), K-step 32; 8 waves in 4x2, each 32x32 (2x2 WMMA)
__global__ __launch_bounds__(256) void moe_ffn13(
    const float* __restrict__ x,       // [TOK][HID]
    const float* __restrict__ w1,      // [FF][HID] expert slice
    const float* __restrict__ w3,      // [FF][HID] expert slice
    const int*   __restrict__ idx,     // [TOK] gathered token ids for this expert
    const int*   __restrict__ cntp,    // &counts[e]
    __bf16* __restrict__ inter)        // [TOK][FF] slot-major bf16
{
    const int cnt   = *cntp;
    const int mbase = blockIdx.x * 128;
    if (mbase >= cnt) return;          // block-uniform early exit

    __shared__ __align__(64) __bf16 lA [128 * 32];
    __shared__ __align__(64) __bf16 lB1[ 64 * 32];
    __shared__ __align__(64) __bf16 lB3[ 64 * 32];

    const int tid  = threadIdx.x;
    const int lane = tid & 31;
    const int wid  = tid >> 5;
    const int wm   = (wid & 3) * 32;
    const int wn   = (wid >> 2) * 32;
    const int half = lane >> 4;
    const int l15  = lane & 15;
    const int nbase = blockIdx.y * 64;

    // precomputed gathered A-row sources: thread owns 2 fixed (row, kchunk) slots
    const int arow0 = tid >> 2;               // 0..63
    const int arow1 = (tid + 256) >> 2;       // 64..127
    const int akc   = (tid & 3) * 8;
    const int s0g = mbase + arow0, s1g = mbase + arow1;
    const int tok0 = (s0g < cnt) ? idx[s0g] : 0;
    const int tok1 = (s1g < cnt) ? idx[s1g] : 0;
    const float* ga0 = x + (size_t)tok0 * HID + akc;
    const float* ga1 = x + (size_t)tok1 * HID + akc;
    // B sources (w row contiguous in K=H)
    const int brow = tid >> 2;                // 0..63
    const float* gb1 = w1 + (size_t)(nbase + brow) * HID + akc;
    const float* gb3 = w3 + (size_t)(nbase + brow) * HID + akc;

    v8f acc1[2][2], acc3[2][2];
#pragma unroll
    for (int mi = 0; mi < 2; ++mi)
#pragma unroll
        for (int ni = 0; ni < 2; ++ni) { acc1[mi][ni] = (v8f)0.0f; acc3[mi][ni] = (v8f)0.0f; }

    for (int k0 = 0; k0 < HID; k0 += 32) {
        __syncthreads();
        { // A rows (gathered), f32 -> bf16
            v4f f0 = *(const v4f*)(ga0 + k0), f1 = *(const v4f*)(ga0 + k0 + 4);
            *(v8bf*)&lA[arow0 * 32 + akc] = cvt8(f0, f1);
            v4f g0 = *(const v4f*)(ga1 + k0), g1 = *(const v4f*)(ga1 + k0 + 4);
            *(v8bf*)&lA[arow1 * 32 + akc] = cvt8(g0, g1);
            if (k0 + 32 < HID) { __builtin_prefetch(ga0 + k0 + 32, 0, 3); __builtin_prefetch(ga1 + k0 + 32, 0, 3); }
        }
        { // B1/B3, f32 -> bf16
            v4f a0 = *(const v4f*)(gb1 + k0), a1 = *(const v4f*)(gb1 + k0 + 4);
            v4f b0 = *(const v4f*)(gb3 + k0), b1 = *(const v4f*)(gb3 + k0 + 4);
            *(v8bf*)&lB1[brow * 32 + akc] = cvt8(a0, a1);
            *(v8bf*)&lB3[brow * 32 + akc] = cvt8(b0, b1);
            if (k0 + 32 < HID) { __builtin_prefetch(gb1 + k0 + 32, 0, 3); __builtin_prefetch(gb3 + k0 + 32, 0, 3); }
        }
        __syncthreads();

        v16bf afr[2];
#pragma unroll
        for (int mi = 0; mi < 2; ++mi) {
            int r = wm + mi * 16 + l15;
            v8bf c0 = *(const v8bf*)&lA[r * 32 + half * 16];
            v8bf c1 = *(const v8bf*)&lA[r * 32 + half * 16 + 8];
#pragma unroll
            for (int i = 0; i < 8; ++i) { afr[mi][i] = c0[i]; afr[mi][i + 8] = c1[i]; }
        }
        v16bf b1fr[2], b3fr[2];
#pragma unroll
        for (int ni = 0; ni < 2; ++ni) {
            int r = wn + ni * 16 + l15;
            b1fr[ni] = *(const v16bf*)&lB1[r * 32 + half * 16];
            b3fr[ni] = *(const v16bf*)&lB3[r * 32 + half * 16];
        }
#pragma unroll
        for (int mi = 0; mi < 2; ++mi)
#pragma unroll
            for (int ni = 0; ni < 2; ++ni) {
                acc1[mi][ni] = __builtin_amdgcn_wmma_f32_16x16x32_bf16(
                    false, afr[mi], false, b1fr[ni], (short)0, acc1[mi][ni], false, false);
                acc3[mi][ni] = __builtin_amdgcn_wmma_f32_16x16x32_bf16(
                    false, afr[mi], false, b3fr[ni], (short)0, acc3[mi][ni], false, false);
            }
    }

    // epilogue: silu(a1)*a3 -> bf16 inter (slot-major rows)
#pragma unroll
    for (int mi = 0; mi < 2; ++mi)
#pragma unroll
        for (int ni = 0; ni < 2; ++ni)
#pragma unroll
            for (int r = 0; r < 8; ++r) {
                int slot = mbase + wm + mi * 16 + half * 8 + r;
                if (slot < cnt) {
                    float a = acc1[mi][ni][r];
                    float b = acc3[mi][ni][r];
                    float s = a * __frcp_rn(1.0f + __expf(-a));   // silu
                    int col = nbase + wn + ni * 16 + l15;
                    inter[(size_t)slot * FF + col] = (__bf16)(s * b);
                }
            }
}

// ---------------- expert GEMM 2 (gathered): out[tok] += gw * (inter @ w2^T) ----------------
__global__ __launch_bounds__(256) void moe_ffn2(
    const __bf16* __restrict__ inter,  // [TOK][FF] slot-major bf16
    const float*  __restrict__ w2,     // [HID][FF] expert slice
    const int*    __restrict__ idx,    // [TOK] gathered token ids
    const float*  __restrict__ gathw,  // [TOK] gathered gate weights
    const int*    __restrict__ cntp,   // &counts[e]
    float* __restrict__ out)           // [TOK][HID] accumulated (experts serialized)
{
    const int cnt   = *cntp;
    const int mbase = blockIdx.x * 128;
    if (mbase >= cnt) return;

    __shared__ __align__(64) __bf16 lA[128 * 32];
    __shared__ __align__(64) __bf16 lB[ 64 * 32];

    const int tid  = threadIdx.x;
    const int lane = tid & 31;
    const int wid  = tid >> 5;
    const int wm   = (wid & 3) * 32;
    const int wn   = (wid >> 2) * 32;
    const int half = lane >> 4;
    const int l15  = lane & 15;
    const int nbase = blockIdx.y * 64;

    const int arow0 = tid >> 2;
    const int arow1 = (tid + 256) >> 2;
    const int akc   = (tid & 3) * 8;
    const __bf16* ga0 = inter + (size_t)(mbase + arow0) * FF + akc;  // slot-major: contiguous
    const __bf16* ga1 = inter + (size_t)(mbase + arow1) * FF + akc;
    const int brow = tid >> 2;
    const float* gb = w2 + (size_t)(nbase + brow) * FF + akc;

    v8f acc[2][2];
#pragma unroll
    for (int mi = 0; mi < 2; ++mi)
#pragma unroll
        for (int ni = 0; ni < 2; ++ni) acc[mi][ni] = (v8f)0.0f;

    for (int k0 = 0; k0 < FF; k0 += 32) {
        __syncthreads();
        // A tile: already bf16 — use CDNA5 async global->LDS copy when available
#if HAVE_ASYNC_B128
        ASYNC_CP_B128(ga0 + k0, &lA[arow0 * 32 + akc]);
        ASYNC_CP_B128(ga1 + k0, &lA[arow1 * 32 + akc]);
#else
        *(v8bf*)&lA[arow0 * 32 + akc] = *(const v8bf*)(ga0 + k0);
        *(v8bf*)&lA[arow1 * 32 + akc] = *(const v8bf*)(ga1 + k0);
#endif
        { // B tile: f32 -> bf16
            v4f f0 = *(const v4f*)(gb + k0), f1 = *(const v4f*)(gb + k0 + 4);
            *(v8bf*)&lB[brow * 32 + akc] = cvt8(f0, f1);
            if (k0 + 32 < FF) __builtin_prefetch(gb + k0 + 32, 0, 3);
        }
#if HAVE_ASYNC_B128
#  if HAVE_WAIT_ASYNC
        __builtin_amdgcn_s_wait_asynccnt(0);
#  else
        asm volatile("s_wait_asynccnt 0" ::: "memory");
#  endif
#endif
        __syncthreads();

        v16bf afr[2];
#pragma unroll
        for (int mi = 0; mi < 2; ++mi) {
            int r = wm + mi * 16 + l15;
            v8bf c0 = *(const v8bf*)&lA[r * 32 + half * 16];
            v8bf c1 = *(const v8bf*)&lA[r * 32 + half * 16 + 8];
#pragma unroll
            for (int i = 0; i < 8; ++i) { afr[mi][i] = c0[i]; afr[mi][i + 8] = c1[i]; }
        }
        v16bf bfr[2];
#pragma unroll
        for (int ni = 0; ni < 2; ++ni) {
            int r = wn + ni * 16 + l15;
            bfr[ni] = *(const v16bf*)&lB[r * 32 + half * 16];
        }
#pragma unroll
        for (int mi = 0; mi < 2; ++mi)
#pragma unroll
            for (int ni = 0; ni < 2; ++ni)
                acc[mi][ni] = __builtin_amdgcn_wmma_f32_16x16x32_bf16(
                    false, afr[mi], false, bfr[ni], (short)0, acc[mi][ni], false, false);
    }

    // epilogue: scatter out[tok] += gw * acc (experts serialized on stream -> RMW safe)
#pragma unroll
    for (int mi = 0; mi < 2; ++mi)
#pragma unroll
        for (int ni = 0; ni < 2; ++ni)
#pragma unroll
            for (int r = 0; r < 8; ++r) {
                int slot = mbase + wm + mi * 16 + half * 8 + r;
                if (slot < cnt) {
                    int tok   = idx[slot];
                    float wgt = gathw[slot];
                    int col   = nbase + wn + ni * 16 + l15;
                    size_t o  = (size_t)tok * HID + col;
                    out[o] += wgt * acc[mi][ni][r];
                }
            }
}

// ---------------- launcher ----------------
extern "C" void kernel_launch(void* const* d_in, const int* in_sizes, int n_in,
                              void* d_out, int out_size, void* d_ws, size_t ws_size,
                              hipStream_t stream)
{
    const float* x  = (const float*)d_in[0];   // [B,S,H]
    const float* gw = (const float*)d_in[1];   // [E,H]
    const float* w1 = (const float*)d_in[2];   // [E,F,H]
    const float* w2 = (const float*)d_in[3];   // [E,H,F]
    const float* w3 = (const float*)d_in[4];   // [E,F,H]
    // d_in[5] = top_k (fixed 2)

    float* out    = (float*)d_out;                 // TOK*HID
    float* logits = out + (size_t)TOK * HID;       // TOK*NE (2nd tuple output)

    char* ws = (char*)d_ws;
    int*   counts = (int*)ws;                                         // NE ints
    size_t o1 = 256;
    int*   idx    = (int*)(ws + o1);                                  // NE*TOK ints
    size_t o2 = o1 + (((size_t)NE * TOK * sizeof(int) + 255) & ~(size_t)255);
    float* gathw  = (float*)(ws + o2);                                // NE*TOK f32
    size_t o3 = o2 + (((size_t)NE * TOK * sizeof(float) + 255) & ~(size_t)255);
    __bf16* inter = (__bf16*)(ws + o3);                               // TOK*FF bf16 (32 MB)

    moe_zero<<<(TOK * HID + 255) / 256, 256, 0, stream>>>(out, TOK * HID);
    moe_zero_counts<<<1, 32, 0, stream>>>(counts);
    moe_router<<<TOK / 8, 256, 0, stream>>>(x, gw, counts, idx, gathw, logits);

    for (int e = 0; e < NE; ++e) {
        const float* w1e = w1 + (size_t)e * FF * HID;
        const float* w3e = w3 + (size_t)e * FF * HID;
        const float* w2e = w2 + (size_t)e * HID * FF;
        const int*   ide = idx + (size_t)e * TOK;
        const float* gwe = gathw + (size_t)e * TOK;
        moe_ffn13<<<dim3(TOK / 128, FF / 64), 256, 0, stream>>>(x, w1e, w3e, ide, counts + e, inter);
        moe_ffn2 <<<dim3(TOK / 128, HID / 64), 256, 0, stream>>>(inter, w2e, ide, gwe, counts + e, out);
    }
}
